// GraphConstructorGDN_12206297055832
// MI455X (gfx1250) — matrix operand order repaired and checked by reading, not compiled, not run on hardware
//
#include <hip/hip_runtime.h>
#include <stdint.h>

#define NNODES 10000
#define DIM    64
#define TOPK   30
#define NTILE  (NNODES / 16)   // 625, exact
#define NBINS  1024            // histogram bins over cos in [-1,1]

typedef __attribute__((ext_vector_type(16))) __bf16         v16bf;
typedef __attribute__((ext_vector_type(8)))  float          v8f;
typedef __attribute__((ext_vector_type(8)))  unsigned short v8us;

union ABu { v16bf v; v8us h[2]; };

__device__ __forceinline__ unsigned short f32_to_bf16_rne(float f) {
    unsigned int u = __float_as_uint(f);
    unsigned int r = u + 0x7FFFu + ((u >> 16) & 1u);
    return (unsigned short)(r >> 16);
}

// ---------------------------------------------------------------------------
// Kernel 0: gather + bf16 convert + reciprocal row norms (fp32).
// w is tiny (1.28 MB bf16) and stays L2-resident for the GEMM passes.
// ---------------------------------------------------------------------------
__global__ void prep_kernel(const int* __restrict__ idx,
                            const float* __restrict__ emb,
                            unsigned short* __restrict__ wbf,
                            float* __restrict__ rnorm) {
    int row = blockIdx.x * blockDim.x + threadIdx.x;
    if (row >= NNODES) return;
    const float* e = emb + (size_t)idx[row] * DIM;
    unsigned short* o = wbf + (size_t)row * DIM;
    float ss = 0.0f;
    #pragma unroll
    for (int k = 0; k < DIM; ++k) {
        float v = e[k];
        ss += v * v;
        o[k] = f32_to_bf16_rne(v);
    }
    rnorm[row] = rsqrtf(ss);
}

// ---------------------------------------------------------------------------
// Shared GEMM tile helper: load A (16x64 rows m0..m0+15) in contiguous k order (k = lh*16 + s), matching B.
// ---------------------------------------------------------------------------
__device__ __forceinline__ void load_A(const unsigned short* __restrict__ wbf,
                                       int m0, int lane, int lh,
                                       ABu& A0, ABu& A1) {
    const unsigned short* ar = wbf + (size_t)(m0 + (lane & 15)) * DIM;
    A0.h[0] = *(const v8us*)(ar + lh * 16);         // K =  lh*16 .. +7
    A0.h[1] = *(const v8us*)(ar + 8 + lh * 16);     // K = lh*16+8 .. +7
    A1.h[0] = *(const v8us*)(ar + 32 + lh * 16);    // +32 K-block
    A1.h[1] = *(const v8us*)(ar + 40 + lh * 16);
}

// ---------------------------------------------------------------------------
// Pass 1: bf16 WMMA GEMM over full rows -> per-row histogram in LDS ->
// per-row threshold = lower edge of bin where cumulative count >= TOPK.
// Histogram packed: two rows per u32 word (counts <= 10000 < 65536, no carry).
// ---------------------------------------------------------------------------
__global__ void __launch_bounds__(256)
topk_thresh_kernel(const unsigned short* __restrict__ wbf,
                   const float* __restrict__ rnorm,
                   float* __restrict__ thresh) {
    __shared__ unsigned int hist[NBINS][8];   // [bin][rowpair], 32 KB
    __shared__ float rm_s[16];

    const int tid  = threadIdx.x;
    const int lane = tid & 31;
    const int wave = tid >> 5;
    const int lh   = lane >> 4;               // lane half: 0 or 1
    const int m0   = blockIdx.x * 16;

    for (int i = tid; i < NBINS * 8; i += 256) ((unsigned int*)hist)[i] = 0u;
    if (tid < 16) rm_s[tid] = rnorm[m0 + tid];
    __syncthreads();

    ABu A0, A1;
    load_A(wbf, m0, lane, lh, A0, A1);

    float rm[8];
    #pragma unroll
    for (int r = 0; r < 8; ++r) rm[r] = rm_s[lh * 8 + r];

    for (int nt = wave; nt < NTILE; nt += 8) {
        const int n = nt * 16 + (lane & 15);
        const unsigned short* br = wbf + (size_t)n * DIM + lh * 16;
        ABu B0, B1;
        B0.h[0] = *(const v8us*)(br);          // K = lh*16 .. +15
        B0.h[1] = *(const v8us*)(br + 8);
        B1.h[0] = *(const v8us*)(br + 32);     // +32 K-block
        B1.h[1] = *(const v8us*)(br + 40);

        v8f c = {};
        c = __builtin_amdgcn_wmma_f32_16x16x32_bf16(false, A0.v, false, B0.v,
                                                    (short)0, c, false, false);
        c = __builtin_amdgcn_wmma_f32_16x16x32_bf16(false, A1.v, false, B1.v,
                                                    (short)0, c, false, false);

        const float rn = rnorm[n];
        #pragma unroll
        for (int r = 0; r < 8; ++r) {
            float cos = c[r] * rm[r] * rn;
            int b = (int)((cos * 0.5f + 0.5f) * (float)NBINS);
            b = b < 0 ? 0 : (b > NBINS - 1 ? NBINS - 1 : b);
            int row = lh * 8 + r;
            atomicAdd(&hist[b][row >> 1], 1u << ((row & 1) * 16));
        }
    }
    __syncthreads();

    if (tid < 16) {
        unsigned int cum = 0;
        float thr = -1.0f;
        const int sh = (tid & 1) * 16;
        for (int b = NBINS - 1; b >= 0; --b) {
            cum += (hist[b][tid >> 1] >> sh) & 0xFFFFu;
            if (cum >= TOPK) { thr = -1.0f + 2.0f * (float)b / (float)NBINS; break; }
        }
        thresh[m0 + tid] = thr;
    }
}

// ---------------------------------------------------------------------------
// Pass 2: recompute the (free) GEMM, write cos if >= row threshold else 0.
// Every output element is written exactly once -> deterministic, no pre-zero.
// ---------------------------------------------------------------------------
__global__ void __launch_bounds__(256)
adj_write_kernel(const unsigned short* __restrict__ wbf,
                 const float* __restrict__ rnorm,
                 const float* __restrict__ thresh,
                 float* __restrict__ out) {
    __shared__ float rm_s[16];
    __shared__ float th_s[16];

    const int tid  = threadIdx.x;
    const int lane = tid & 31;
    const int wave = tid >> 5;
    const int lh   = lane >> 4;
    const int m0   = blockIdx.x * 16;

    if (tid < 16) { rm_s[tid] = rnorm[m0 + tid]; th_s[tid] = thresh[m0 + tid]; }
    __syncthreads();

    ABu A0, A1;
    load_A(wbf, m0, lane, lh, A0, A1);

    float rm[8], th[8];
    #pragma unroll
    for (int r = 0; r < 8; ++r) { rm[r] = rm_s[lh * 8 + r]; th[r] = th_s[lh * 8 + r]; }

    for (int nt = wave; nt < NTILE; nt += 8) {
        const int n = nt * 16 + (lane & 15);
        const unsigned short* br = wbf + (size_t)n * DIM + lh * 16;
        ABu B0, B1;
        B0.h[0] = *(const v8us*)(br);
        B0.h[1] = *(const v8us*)(br + 8);
        B1.h[0] = *(const v8us*)(br + 32);
        B1.h[1] = *(const v8us*)(br + 40);

        v8f c = {};
        c = __builtin_amdgcn_wmma_f32_16x16x32_bf16(false, A0.v, false, B0.v,
                                                    (short)0, c, false, false);
        c = __builtin_amdgcn_wmma_f32_16x16x32_bf16(false, A1.v, false, B1.v,
                                                    (short)0, c, false, false);

        const float rn = rnorm[n];
        #pragma unroll
        for (int r = 0; r < 8; ++r) {
            float cos = c[r] * rm[r] * rn;
            int row = m0 + lh * 8 + r;
            out[(size_t)row * NNODES + n] = (cos >= th[r]) ? cos : 0.0f;
        }
    }
}

// ---------------------------------------------------------------------------
extern "C" void kernel_launch(void* const* d_in, const int* in_sizes, int n_in,
                              void* d_out, int out_size, void* d_ws, size_t ws_size,
                              hipStream_t stream) {
    const int*   idx = (const int*)d_in[0];
    const float* emb = (const float*)d_in[1];

    unsigned short* wbf   = (unsigned short*)d_ws;                       // 1.28 MB
    float*          rnorm = (float*)((char*)d_ws + (size_t)NNODES * DIM * 2);
    float*          thr   = rnorm + NNODES;
    float*          out   = (float*)d_out;

    prep_kernel<<<(NNODES + 127) / 128, 128, 0, stream>>>(idx, emb, wbf, rnorm);
    topk_thresh_kernel<<<NTILE, 256, 0, stream>>>(wbf, rnorm, thr);
    adj_write_kernel<<<NTILE, 256, 0, stream>>>(wbf, rnorm, thr, out);
}
